// CondConv2d_47236050321814
// MI455X (gfx1250) — matrix operand order repaired and checked by reading, not compiled, not run on hardware
//
#include <hip/hip_runtime.h>

// ---------- types ----------
typedef __attribute__((ext_vector_type(16))) __bf16        bf16x16;
typedef __attribute__((ext_vector_type(8)))  float         f32x8;
typedef __attribute__((ext_vector_type(4)))  unsigned int  u32x4;

union FragBF { bf16x16 v; u32x4 q[2]; };
union Pack8  { u32x4 q; unsigned short s[8]; };

__device__ __forceinline__ unsigned short f32_to_bf16(float f) {
    unsigned int u = __float_as_uint(f);
    unsigned int r = u + 0x7FFFu + ((u >> 16) & 1u);   // round-to-nearest-even
    return (unsigned short)(r >> 16);
}

// gfx1250 async global->LDS (ASYNCcnt path), 16B per lane
__device__ __forceinline__ void async_load_b128(unsigned lds_addr, const void* gaddr) {
    asm volatile("global_load_async_to_lds_b128 %0, %1, off"
                 :: "v"(lds_addr), "v"((unsigned long long)(uintptr_t)gaddr)
                 : "memory");
}
__device__ __forceinline__ void wait_async0() {
    asm volatile("s_wait_asynccnt 0x0" ::: "memory");
}
__device__ __forceinline__ unsigned lds_addr_of(const void* p) {
    // generic shared-aperture address: low 32 bits are the LDS byte offset
    return (unsigned)(uintptr_t)p;
}

// ---------- problem constants ----------
#define BATCH 32
#define CIN   256
#define COUT  256
#define HIMG  56
#define WIMG  56
#define HW    3136      // 56*56
#define KDIM  2304      // 9*CIN  (k ordered as [ki*3+kj][c])
#define WPB   589824    // COUT*KDIM per batch sample
#define KTILE 64
#define NIT   (KDIM / KTILE)   // 36

// ================== kernel 1: global average pool ==================
__global__ __launch_bounds__(256) void pool_kernel(const float* __restrict__ x,
                                                   float* __restrict__ pooled) {
    __shared__ float red[256];
    const int bc = blockIdx.x;
    const float* p = x + (size_t)bc * HW;
    float s = 0.f;
    for (int i = threadIdx.x; i < HW; i += 256) s += p[i];
    red[threadIdx.x] = s;
    __syncthreads();
    for (int off = 128; off > 0; off >>= 1) {
        if (threadIdx.x < off) red[threadIdx.x] += red[threadIdx.x + off];
        __syncthreads();
    }
    if (threadIdx.x == 0) pooled[bc] = red[0] * (1.0f / (float)HW);
}

// ================== kernel 2: routing MLP + softmax ==================
__global__ __launch_bounds__(64) void route_kernel(const float* __restrict__ pooled,
                                                   const float* __restrict__ rw1,
                                                   const float* __restrict__ rb1,
                                                   const float* __restrict__ rw2,
                                                   const float* __restrict__ rb2,
                                                   float* __restrict__ routing) {
    __shared__ float h[64];
    __shared__ float lg[4];
    const int b = blockIdx.x, t = threadIdx.x;
    const float* pb = pooled + b * CIN;
    float s = rb1[t];
    const float* w = rw1 + t * CIN;
    for (int i = 0; i < CIN; ++i) s += pb[i] * w[i];
    h[t] = fmaxf(s, 0.f);
    __syncthreads();
    if (t < 4) {
        float l = rb2[t];
        const float* w2 = rw2 + t * 64;
        for (int j = 0; j < 64; ++j) l += h[j] * w2[j];
        lg[t] = l;
    }
    __syncthreads();
    if (t == 0) {
        float m = fmaxf(fmaxf(lg[0], lg[1]), fmaxf(lg[2], lg[3]));
        float e0 = __expf(lg[0] - m), e1 = __expf(lg[1] - m);
        float e2 = __expf(lg[2] - m), e3 = __expf(lg[3] - m);
        float inv = 1.f / (e0 + e1 + e2 + e3);
        routing[b * 4 + 0] = e0 * inv;
        routing[b * 4 + 1] = e1 * inv;
        routing[b * 4 + 2] = e2 * inv;
        routing[b * 4 + 3] = e3 * inv;
    }
}

// ========== kernel 3: mix experts, permute k to [9][C], cast bf16 ==========
// wcomb[b][o][r9*256 + c] = sum_e routing[b][e] * experts[e][o][c*9 + r9]
__global__ __launch_bounds__(256) void combine_kernel(const float* __restrict__ experts,
                                                      const float* __restrict__ routing,
                                                      unsigned short* __restrict__ wcomb) {
    long long i = (long long)blockIdx.x * 256 + threadIdx.x;
    if (i >= (long long)BATCH * WPB) return;
    int b = (int)(i / WPB);
    int t = (int)(i - (long long)b * WPB);
    int o  = t / KDIM;
    int u  = t - o * KDIM;
    int r9 = u >> 8;        // 0..8
    int c  = u & 255;
    int src = o * KDIM + c * 9 + r9;
    float r0 = routing[b * 4 + 0], r1 = routing[b * 4 + 1];
    float r2 = routing[b * 4 + 2], r3 = routing[b * 4 + 3];
    float v = r0 * experts[src]
            + r1 * experts[WPB + src]
            + r2 * experts[2 * WPB + src]
            + r3 * experts[3 * WPB + src];
    wcomb[i] = f32_to_bf16(v);
}

// ================== kernel 4: implicit-GEMM conv via WMMA bf16 ==================
// grid: x = 49 (N tiles of 64 px), y = 32 (batch). Block tile 256(M) x 64(N).
// 256 threads = 8 wave32 arranged 4(M) x 2(N); each wave owns a 64x32 output tile
// (8 accumulators, 16 WMMAs per LDS stage). Double-buffered LDS (80 KB);
// A tile via async global->LDS (ASYNCcnt), B tile im2col'd with immediate offsets.
__global__ __launch_bounds__(256) void condconv_wmma_kernel(
        const float* __restrict__ x,
        const unsigned short* __restrict__ wcomb,
        float* __restrict__ out) {
    __shared__ alignas(16) unsigned short Alds[2][256 * KTILE];  // 2 x 32 KB
    __shared__ alignas(16) unsigned short Blds[2][64 * KTILE];   // 2 x  8 KB

    const int tid  = threadIdx.x;
    const int lane = tid & 31;
    const int half = lane >> 4;
    const int l16  = lane & 15;
    const int wave = tid >> 5;
    const int wm   = (wave >> 1) * 64;   // 0,64,128,192
    const int wn   = (wave & 1) * 32;    // 0,32

    const int n0 = blockIdx.x * 64;
    const int b  = blockIdx.y;

    const unsigned short* wbase = wcomb + (size_t)b * WPB;     // full M=256 per block
    const float* xb = x + (size_t)b * CIN * HW;

    // B-loader coords: thread -> pixel n, 16 consecutive channels
    const int bn  = tid & 63;
    const int bkl = (tid >> 6) * 16;            // 0,16,32,48
    const int ohw = n0 + bn;
    const int oh  = ohw / WIMG;
    const int ow  = ohw - oh * WIMG;

    const unsigned aAddr0 = lds_addr_of(&Alds[0][0]);
    const unsigned aAddr1 = lds_addr_of(&Alds[1][0]);

    f32x8 acc[4][2];
    #pragma unroll
    for (int mi = 0; mi < 4; ++mi)
        #pragma unroll
        for (int ni = 0; ni < 2; ++ni)
            acc[mi][ni] = (f32x8){0.f,0.f,0.f,0.f,0.f,0.f,0.f,0.f};

    // ---- issue A-tile async loads for a stage: 256 x 64 bf16, 8 x 16B per thread
    auto issue_A = [&](int kt, unsigned aBase) {
        #pragma unroll
        for (int c8 = 0; c8 < 8; ++c8) {
            int ch  = tid + c8 * 256;            // 0..2047
            int row = ch >> 3;                   // 0..255
            int col = (ch & 7) * 8;              // 0..56
            async_load_b128(aBase + (unsigned)(row * (KTILE * 2) + col * 2),
                            wbase + (size_t)row * KDIM + kt + col);
        }
    };
    // ---- B im2col: 16 global f32 loads into regs (fixed (di,dj) per tile)
    auto load_B = [&](int kt, float* bv) {
        int r9 = kt >> 8;                        // plane index 0..8 (uniform)
        int di = r9 / 3, dj = r9 - di * 3;
        int ih = oh + di - 1, iw = ow + dj - 1;
        bool ok = ((unsigned)ih < (unsigned)HIMG) && ((unsigned)iw < (unsigned)WIMG);
        int cb = (kt & 255) + bkl;               // channel base for this thread
        const float* bp = xb + (size_t)cb * HW + ih * WIMG + iw;
        #pragma unroll
        for (int j = 0; j < 16; ++j) bv[j] = ok ? bp[(size_t)j * HW] : 0.f;
    };
    auto store_B = [&](unsigned short* Bbuf, const float* bv) {
        Pack8 p0, p1;
        #pragma unroll
        for (int j = 0; j < 8; ++j) { p0.s[j] = f32_to_bf16(bv[j]); p1.s[j] = f32_to_bf16(bv[8 + j]); }
        *(u32x4*)(&Bbuf[bn * KTILE + bkl])     = p0.q;
        *(u32x4*)(&Bbuf[bn * KTILE + bkl + 8]) = p1.q;
    };

    // ---- prologue: stage 0 into buffer 0
    issue_A(0, aAddr0);
    {
        float bv[16];
        load_B(0, bv);
        store_B(Blds[0], bv);
    }
    wait_async0();
    __syncthreads();

    for (int it = 0; it < NIT; ++it) {
        const int buf = it & 1;
        const unsigned short* Abuf = Alds[buf];
        const unsigned short* Bbuf = Blds[buf];
        const bool havenext = (it + 1) < NIT;    // uniform branch

        float bv[16];
        if (havenext) {
            const int nkt = (it + 1) * KTILE;
            issue_A(nkt, buf ? aAddr0 : aAddr1); // next buffer
            load_B(nkt, bv);                     // global loads stay in flight
        }

        // ---- 2 k-steps x 8 WMMAs on current buffers
        #pragma unroll
        for (int ks = 0; ks < KTILE; ks += 32) {
            FragBF fa[4], fb[2];
            #pragma unroll
            for (int mi = 0; mi < 4; ++mi) {
                const unsigned short* ar = &Abuf[(wm + mi * 16 + l16) * KTILE + ks + half * 8];
                fa[mi].q[0] = *(const u32x4*)ar;
                fa[mi].q[1] = *(const u32x4*)(ar + 16);
            }
            #pragma unroll
            for (int ni = 0; ni < 2; ++ni) {
                const unsigned short* br = &Bbuf[(wn + ni * 16 + l16) * KTILE + ks + half * 8];
                fb[ni].q[0] = *(const u32x4*)br;
                fb[ni].q[1] = *(const u32x4*)(br + 16);
            }
            #pragma unroll
            for (int mi = 0; mi < 4; ++mi)
                #pragma unroll
                for (int ni = 0; ni < 2; ++ni)
                    acc[mi][ni] = __builtin_amdgcn_wmma_f32_16x16x32_bf16(
                        false, fa[mi].v, false, fb[ni].v, (short)0, acc[mi][ni], false, false);
        }

        if (havenext) store_B(Blds[buf ^ 1], bv);
        wait_async0();
        __syncthreads();
    }

    // ---- epilogue: C/D layout VGPR r -> M = r + 8*half, col = lane&15
    float* ob = out + ((size_t)b * COUT + wm) * HW + n0 + wn;
    #pragma unroll
    for (int mi = 0; mi < 4; ++mi) {
        #pragma unroll
        for (int r = 0; r < 8; ++r) {
            int m = mi * 16 + r + 8 * half;
            #pragma unroll
            for (int ni = 0; ni < 2; ++ni)
                ob[(size_t)m * HW + ni * 16 + l16] = acc[mi][ni][r];
        }
    }
}

// ================== launcher ==================
extern "C" void kernel_launch(void* const* d_in, const int* in_sizes, int n_in,
                              void* d_out, int out_size, void* d_ws, size_t ws_size,
                              hipStream_t stream) {
    const float* x       = (const float*)d_in[0];
    const float* experts = (const float*)d_in[1];
    const float* rw1     = (const float*)d_in[2];
    const float* rb1     = (const float*)d_in[3];
    const float* rw2     = (const float*)d_in[4];
    const float* rb2     = (const float*)d_in[5];
    float* out = (float*)d_out;

    char* ws = (char*)d_ws;
    float* pooled         = (float*)ws;                    // 32 KB
    float* routing        = (float*)(ws + 32768);          // 512 B
    unsigned short* wcomb = (unsigned short*)(ws + 33280); // ~36 MB

    pool_kernel<<<dim3(BATCH * CIN), dim3(256), 0, stream>>>(x, pooled);
    route_kernel<<<dim3(BATCH), dim3(64), 0, stream>>>(pooled, rw1, rb1, rw2, rb2, routing);

    long long nmix = (long long)BATCH * WPB;
    combine_kernel<<<dim3((unsigned)((nmix + 255) / 256)), dim3(256), 0, stream>>>(experts, routing, wcomb);

    condconv_wmma_kernel<<<dim3(49, BATCH), dim3(256), 0, stream>>>(x, wcomb, out);
}
